// mpnn_82729660056231
// MI455X (gfx1250) — matrix operand order repaired and checked by reading, not compiled, not run on hardware
//
#include <hip/hip_runtime.h>
#include <hip/hip_bf16.h>

// ---------------------------------------------------------------------------
// MPNN for MI455X (gfx1250): WMMA bf16 GEMMs + f32 atomic graph aggregation.
// ---------------------------------------------------------------------------

typedef __attribute__((ext_vector_type(16))) __bf16        v16bf;
typedef __attribute__((ext_vector_type(8)))  float         v8f;
typedef __attribute__((ext_vector_type(4)))  unsigned int  u32x4;
typedef __attribute__((ext_vector_type(8)))  unsigned int  u32x8;

static __device__ __forceinline__ unsigned short f32_to_bf16_rne(float f) {
    union { float f; unsigned int u; } c; c.f = f;
    unsigned int u = c.u;
    u += 0x7FFFu + ((u >> 16) & 1u);   // round-to-nearest-even
    return (unsigned short)(u >> 16);
}

// ---------------- small elementwise / prep kernels -------------------------

__global__ void mpnn_zero_f32(float* __restrict__ p, size_t n) {
    size_t i = (size_t)blockIdx.x * blockDim.x + threadIdx.x;
    if (i < n) p[i] = 0.0f;
}

__global__ void mpnn_cvt_bf16(const float* __restrict__ in,
                              unsigned short* __restrict__ out, int n) {
    int i = blockIdx.x * blockDim.x + threadIdx.x;
    if (i < n) out[i] = f32_to_bf16_rne(in[i]);
}

// in: [R][C] row-major -> out: [C][R] (bf16)
__global__ void mpnn_transpose_cvt_bf16(const float* __restrict__ in,
                                        unsigned short* __restrict__ out,
                                        int R, int C) {
    int i = blockIdx.x * blockDim.x + threadIdx.x;
    if (i >= R * C) return;
    int r = i / C, c = i - r * C;
    out[c * R + r] = f32_to_bf16_rne(in[i]);
}

__global__ void mpnn_degree_count(const int* __restrict__ src,
                                  const int* __restrict__ dst,
                                  float* __restrict__ deg_out,
                                  float* __restrict__ deg_in, int nE) {
    int e = blockIdx.x * blockDim.x + threadIdx.x;
    if (e >= nE) return;
    atomicAdd(&deg_out[src[e]], 1.0f);
    atomicAdd(&deg_in[dst[e]], 1.0f);
}

__global__ void mpnn_finalize_norm(float* __restrict__ deg, int n) {
    int i = blockIdx.x * blockDim.x + threadIdx.x;
    if (i < n) deg[i] = rsqrtf(fmaxf(deg[i], 1.0f));
}

// one wave32 per edge: 4 floats / lane, scaled gather + atomic scatter-add
__global__ void mpnn_edge_agg(const float* __restrict__ h,
                              const float* __restrict__ norm_src,
                              const int* __restrict__ src,
                              const int* __restrict__ dst,
                              float* __restrict__ agg, int nE) {
    size_t tid = (size_t)blockIdx.x * blockDim.x + threadIdx.x;
    size_t e = tid >> 5;
    int l = (int)(tid & 31);
    if (e >= (size_t)nE) return;
    int s = src[e], d = dst[e];
    float ns = norm_src[s];
    const float4 v = *(const float4*)(h + (size_t)s * 128 + l * 4);
    float* o = agg + (size_t)d * 128 + l * 4;
    atomicAdd(o + 0, v.x * ns);
    atomicAdd(o + 1, v.y * ns);
    atomicAdd(o + 2, v.z * ns);
    atomicAdd(o + 3, v.w * ns);
}

__global__ void mpnn_gru_update(const float* __restrict__ gi,
                                const float* __restrict__ gh,
                                float* __restrict__ h, int nN) {
    size_t i = (size_t)blockIdx.x * blockDim.x + threadIdx.x;
    if (i >= (size_t)nN * 128) return;
    size_t node = i >> 7;
    int f = (int)(i & 127);
    const float* gir = gi + node * 384;
    const float* ghr = gh + node * 384;
    float r = 1.0f / (1.0f + __expf(-(gir[f] + ghr[f])));
    float z = 1.0f / (1.0f + __expf(-(gir[128 + f] + ghr[128 + f])));
    float c = tanhf(gir[256 + f] + r * ghr[256 + f]);
    h[i] = (1.0f - z) * c + z * h[i];
}

__global__ void mpnn_pool_sum(const float* __restrict__ a,
                              const float* __restrict__ b,
                              const int* __restrict__ gid,
                              float* __restrict__ out, int nN) {
    size_t i = (size_t)blockIdx.x * blockDim.x + threadIdx.x;
    if (i >= (size_t)nN * 128) return;
    int node = (int)(i >> 7), f = (int)(i & 127);
    atomicAdd(&out[gid[node] * 128 + f], a[i] + b[i]);
}

// ---------------- WMMA GEMM ------------------------------------------------
// C[M,N] = act(A[M,K] (optionally row-scaled) x Bt^T + bias)
// Bt is bf16, N-major / K-contiguous: Bt[n*K + k].
// Block: 128 threads = 4 waves; block tile 16 rows x 128 cols; each wave two
// 16x16 accumulators. Requires M%16==0, N%128==0, K%32==0 (true here).
__global__ __launch_bounds__(128)
void mpnn_wmma_gemm_bf16(const float* __restrict__ A,
                         const unsigned short* __restrict__ Bt,
                         const float* __restrict__ bias,
                         const float* __restrict__ rowscale,
                         float* __restrict__ C,
                         int N, int K, int do_relu) {
    extern __shared__ unsigned short aTile[];   // [16][K] bf16
    const int tid  = threadIdx.x;
    const int wave = tid >> 5;
    const int lane = tid & 31;
    const int m0 = blockIdx.x << 4;
    const int n0 = (blockIdx.y << 7) + (wave << 5);

    // stage A tile (16 x K) as bf16 into LDS, folding optional row scale
    const int elems = K << 4;                   // multiple of 128 -> uniform loop
    for (int idx = tid; idx < elems; idx += 128) {
        const int r = idx / K;
        const int k = idx - r * K;
        const int row = m0 + r;
        float v = A[(size_t)row * K + k];
        if (rowscale) v *= rowscale[row];
        aTile[idx] = f32_to_bf16_rne(v);
    }
    __syncthreads();

    const int hl  = lane >> 4;                  // half: 0 for lanes 0-15, 1 for 16-31
    const int l16 = lane & 15;
    v8f acc0 = {0.f,0.f,0.f,0.f,0.f,0.f,0.f,0.f};
    v8f acc1 = {0.f,0.f,0.f,0.f,0.f,0.f,0.f,0.f};

    const unsigned short* arow  = aTile + l16 * K;                 // A row M=l16
    const unsigned short* brow0 = Bt + (size_t)(n0 + l16) * K;     // B col n0+l16
    const unsigned short* brow1 = brow0 + (size_t)16 * K;          // B col +16

    for (int kb = 0; kb < K; kb += 32) {
        // A fragment (ISA 16-bit 16x32 layout): elements 0..7 = K kb+hl*8..+8,
        // elements 8..15 = K kb+16+hl*8..+8  -> two contiguous 16B LDS reads
        u32x4 alo = *(const u32x4*)(arow + kb + (hl << 3));
        u32x4 ahi = *(const u32x4*)(arow + kb + 16 + (hl << 3));
        u32x8 au = {alo.x, alo.y, alo.z, alo.w, ahi.x, ahi.y, ahi.z, ahi.w};
        v16bf a = __builtin_bit_cast(v16bf, au);
        // B fragment: same K order as A (elements 0..7 = K kb+hl*8.., 8..15 = K kb+16+hl*8..)
        u32x4 b0lo = *(const u32x4*)(brow0 + kb + (hl << 3));
        u32x4 b0hi = *(const u32x4*)(brow0 + kb + 16 + (hl << 3));
        u32x8 b0u = {b0lo.x, b0lo.y, b0lo.z, b0lo.w, b0hi.x, b0hi.y, b0hi.z, b0hi.w};
        v16bf b0 = __builtin_bit_cast(v16bf, b0u);
        u32x4 b1lo = *(const u32x4*)(brow1 + kb + (hl << 3));
        u32x4 b1hi = *(const u32x4*)(brow1 + kb + 16 + (hl << 3));
        u32x8 b1u = {b1lo.x, b1lo.y, b1lo.z, b1lo.w, b1hi.x, b1hi.y, b1hi.z, b1hi.w};
        v16bf b1 = __builtin_bit_cast(v16bf, b1u);
        acc0 = __builtin_amdgcn_wmma_f32_16x16x32_bf16(false, a, false, b0,
                                                       (short)0, acc0, false, false);
        acc1 = __builtin_amdgcn_wmma_f32_16x16x32_bf16(false, a, false, b1,
                                                       (short)0, acc1, false, false);
    }

    // epilogue: C/D layout -> lane half selects M group, VGPR r = row offset
    const int col0 = n0 + l16;
    const int col1 = col0 + 16;
    const float bv0 = bias ? bias[col0] : 0.0f;
    const float bv1 = bias ? bias[col1] : 0.0f;
#pragma unroll
    for (int r = 0; r < 8; ++r) {
        const int row = m0 + (hl << 3) + r;
        float v0 = acc0[r] + bv0;
        float v1 = acc1[r] + bv1;
        if (do_relu) { v0 = fmaxf(v0, 0.0f); v1 = fmaxf(v1, 0.0f); }
        C[(size_t)row * N + col0] = v0;
        C[(size_t)row * N + col1] = v1;
    }
}

// ---------------- host orchestration ---------------------------------------

extern "C" void kernel_launch(void* const* d_in, const int* in_sizes, int n_in,
                              void* d_out, int out_size, void* d_ws, size_t ws_size,
                              hipStream_t stream) {
    (void)n_in; (void)out_size; (void)ws_size;
    const float* feat = (const float*)d_in[0];
    const float* W_gc = (const float*)d_in[1];
    const float* b_gc = (const float*)d_in[2];
    const float* W_ih = (const float*)d_in[3];
    const float* W_hh = (const float*)d_in[4];
    const float* b_ih = (const float*)d_in[5];
    const float* b_hh = (const float*)d_in[6];
    const float* W1   = (const float*)d_in[7];
    const float* b1   = (const float*)d_in[8];
    const float* W2   = (const float*)d_in[9];
    const float* b2   = (const float*)d_in[10];
    const int* esrc = (const int*)d_in[11];
    const int* edst = (const int*)d_in[12];
    const int* gid  = (const int*)d_in[13];
    const int nN = in_sizes[0] / 128;     // 50000
    const int nE = in_sizes[11];          // 1600000
    const int n_layers = 4;               // static in reference

    char* ws = (char*)d_ws;
    size_t off = 0;
    auto carve = [&](size_t bytes) -> char* {
        char* p = ws + off;
        off = (off + bytes + 255) & ~(size_t)255;
        return p;
    };
    float* h    = (float*)carve((size_t)nN * 128 * 4);
    float* agg  = (float*)carve((size_t)nN * 128 * 4);
    float* m    = (float*)carve((size_t)nN * 256 * 4);
    float* gi   = (float*)carve((size_t)nN * 384 * 4);
    float* gh   = (float*)carve((size_t)nN * 384 * 4);
    float* nsrc = (float*)carve((size_t)nN * 4);
    float* ndst = (float*)carve((size_t)nN * 4);
    unsigned short* wgc_t = (unsigned short*)carve(256 * 128 * 2); // [hid][in]
    unsigned short* wih_b = (unsigned short*)carve(384 * 256 * 2); // W_ih row-major
    unsigned short* whh_b = (unsigned short*)carve(384 * 128 * 2);
    unsigned short* w1_b  = (unsigned short*)carve(128 * 128 * 2);
    unsigned short* w2_b  = (unsigned short*)carve(128 * 128 * 2);

    // --- weight prep: bf16 conversion (N-major/K-contig layouts) ---
    mpnn_transpose_cvt_bf16<<<(128 * 256 + 255) / 256, 256, 0, stream>>>(W_gc, wgc_t, 128, 256);
    mpnn_cvt_bf16<<<(384 * 256 + 255) / 256, 256, 0, stream>>>(W_ih, wih_b, 384 * 256);
    mpnn_cvt_bf16<<<(384 * 128 + 255) / 256, 256, 0, stream>>>(W_hh, whh_b, 384 * 128);
    mpnn_cvt_bf16<<<(128 * 128 + 255) / 256, 256, 0, stream>>>(W1, w1_b, 128 * 128);
    mpnn_cvt_bf16<<<(128 * 128 + 255) / 256, 256, 0, stream>>>(W2, w2_b, 128 * 128);

    // --- degree norms ---
    mpnn_zero_f32<<<(nN + 255) / 256, 256, 0, stream>>>(nsrc, (size_t)nN);
    mpnn_zero_f32<<<(nN + 255) / 256, 256, 0, stream>>>(ndst, (size_t)nN);
    mpnn_degree_count<<<(nE + 255) / 256, 256, 0, stream>>>(esrc, edst, nsrc, ndst, nE);
    mpnn_finalize_norm<<<(nN + 255) / 256, 256, 0, stream>>>(nsrc, nN);
    mpnn_finalize_norm<<<(nN + 255) / 256, 256, 0, stream>>>(ndst, nN);

    // h = feat
    hipMemcpyAsync(h, feat, (size_t)nN * 128 * 4, hipMemcpyDeviceToDevice, stream);

    const unsigned mtiles = (unsigned)(nN / 16);        // 3125
    const unsigned nf_blocks = (unsigned)(((size_t)nN * 128 + 255) / 256);

    for (int L = 0; L < n_layers; ++L) {
        // agg = segment_sum((h*norm_src)[src] at dst)
        mpnn_zero_f32<<<nf_blocks, 256, 0, stream>>>(agg, (size_t)nN * 128);
        {
            size_t t = (size_t)nE * 32;
            mpnn_edge_agg<<<(unsigned)((t + 255) / 256), 256, 0, stream>>>(
                h, nsrc, esrc, edst, agg, nE);
        }
        // m = relu((agg*norm_dst) @ W_gc + b_gc)   [N,256]
        mpnn_wmma_gemm_bf16<<<dim3(mtiles, 2), 128, 16 * 128 * 2, stream>>>(
            agg, wgc_t, b_gc, ndst, m, 256, 128, 1);
        // gi = m @ W_ih.T + b_ih                    [N,384]
        mpnn_wmma_gemm_bf16<<<dim3(mtiles, 3), 128, 16 * 256 * 2, stream>>>(
            m, wih_b, b_ih, nullptr, gi, 384, 256, 0);
        // gh = h @ W_hh.T + b_hh                    [N,384]
        mpnn_wmma_gemm_bf16<<<dim3(mtiles, 3), 128, 16 * 128 * 2, stream>>>(
            h, whh_b, b_hh, nullptr, gh, 384, 128, 0);
        // GRU gate fusion, h updated in place
        mpnn_gru_update<<<nf_blocks, 256, 0, stream>>>(gi, gh, h, nN);
    }

    // MLP on feat -> gi ; MLP on h -> gh  (agg reused as hidden buffer)
    mpnn_wmma_gemm_bf16<<<dim3(mtiles, 1), 128, 16 * 128 * 2, stream>>>(
        feat, w1_b, b1, nullptr, agg, 128, 128, 1);
    mpnn_wmma_gemm_bf16<<<dim3(mtiles, 1), 128, 16 * 128 * 2, stream>>>(
        agg, w2_b, b2, nullptr, gi, 128, 128, 0);
    mpnn_wmma_gemm_bf16<<<dim3(mtiles, 1), 128, 16 * 128 * 2, stream>>>(
        h, w1_b, b1, nullptr, agg, 128, 128, 1);
    mpnn_wmma_gemm_bf16<<<dim3(mtiles, 1), 128, 16 * 128 * 2, stream>>>(
        agg, w2_b, b2, nullptr, gh, 128, 128, 0);

    // pooled output: out[gid] += mlp(feat) + mlp(h)
    mpnn_zero_f32<<<(2048 + 255) / 256, 256, 0, stream>>>((float*)d_out, 2048);
    mpnn_pool_sum<<<nf_blocks, 256, 0, stream>>>(gi, gh, gid, (float*)d_out, nN);
}
